// SwinMamba_83760452206812
// MI455X (gfx1250) — matrix operand order repaired and checked by reading, not compiled, not run on hardware
//
#include <hip/hip_runtime.h>
#include <hip/hip_bf16.h>
#include <cmath>

// ---------------------------------------------------------------------------
// SwinMamba forward for MI455X (gfx1250, wave32, WMMA).
// All GEMMs: f16 inputs (packed, K padded to 32) x f16 weights, f32 accumulate
// via v_wmma_f32_16x16x32_f16. Elementwise / scan stay f32.
// GEMM kernel is branchless in the K loop (clamped row addressing) and
// computes a 16x32 output slab per wave (2 WMMAs per K step, shared A frag).
// ---------------------------------------------------------------------------

typedef __attribute__((ext_vector_type(16))) _Float16 v16h;
typedef __attribute__((ext_vector_type(8)))  _Float16 v8h;
typedef __attribute__((ext_vector_type(8)))  float    v8f;

#define DEVINL __device__ __forceinline__

DEVINL float siluf(float v) { return v / (1.0f + __expf(-v)); }
DEVINL int   imin(int a, int b) { return a < b ? a : b; }

// ---------------------------------------------------------------------------
// WMMA GEMM: C[M,N] = A[M,Kp](f16) * W[N,Kp](f16)^T + bias[N]
// One wave computes a 16(M) x 32(N) slab: two 16x16 WMMA tiles sharing the
// A fragment. Out-of-range rows are clamped (loads always valid, EXEC full);
// stores are bounds-checked. Block = 4 waves on 4 consecutive M tiles.
// ---------------------------------------------------------------------------
__global__ __launch_bounds__(128)
void k_gemm_wmma(const _Float16* __restrict__ A, const _Float16* __restrict__ W,
                 const float* __restrict__ bias, float* __restrict__ C,
                 int M, int N, int Kp, long outStride, long outOff)
{
    const int lane = threadIdx.x & 31;
    const int wave = threadIdx.x >> 5;
    const int tm   = blockIdx.y * 4 + wave;   // M tile index (16 rows)
    const int tn   = blockIdx.x;              // N slab index (32 cols)
    if (tm * 16 >= M) return;                 // wave-uniform exit (EXEC stays full)

    const int m  = tm * 16 + (lane & 15);
    const int n0 = tn * 32 + (lane & 15);
    const int n1 = n0 + 16;
    const int kb = (lane < 16) ? 0 : 16;      // A-fragment K sub-offset (contiguous)
    const int ks = (lane < 16) ? 0 : 16;      // B-fragment K sub-offset

    // clamp to valid rows: clamped lanes only feed D elements we never store
    const _Float16* arow  = A + (size_t)imin(m,  M - 1) * Kp;
    const _Float16* wrow0 = W + (size_t)imin(n0, N - 1) * Kp;
    const _Float16* wrow1 = W + (size_t)imin(n1, N - 1) * Kp;

    v8f acc0 = {}, acc1 = {};
    for (int k0 = 0; k0 < Kp; k0 += 32) {
        v8h a0 = *(const v8h*)(arow  + k0 + kb);       // K = kb..kb+7
        v8h a1 = *(const v8h*)(arow  + k0 + kb + 8);   // K = kb+8..kb+15
        v8h p0 = *(const v8h*)(wrow0 + k0 + ks);       // K = ks..ks+7
        v8h p1 = *(const v8h*)(wrow0 + k0 + ks + 8);   // K = ks+8..ks+15
        v8h q0 = *(const v8h*)(wrow1 + k0 + ks);
        v8h q1 = *(const v8h*)(wrow1 + k0 + ks + 8);
        v16h a  = __builtin_shufflevector(a0, a1, 0,1,2,3,4,5,6,7,8,9,10,11,12,13,14,15);
        v16h b0 = __builtin_shufflevector(p0, p1, 0,1,2,3,4,5,6,7,8,9,10,11,12,13,14,15);
        v16h b1 = __builtin_shufflevector(q0, q1, 0,1,2,3,4,5,6,7,8,9,10,11,12,13,14,15);
        acc0 = __builtin_amdgcn_wmma_f32_16x16x32_f16(false, a, false, b0,
                                                      (short)0, acc0, false, false);
        acc1 = __builtin_amdgcn_wmma_f32_16x16x32_f16(false, a, false, b1,
                                                      (short)0, acc1, false, false);
    }

    const int mb = (lane < 16) ? 0 : 8;
    const float bv0 = (bias != nullptr && n0 < N) ? bias[n0] : 0.0f;
    const float bv1 = (bias != nullptr && n1 < N) ? bias[n1] : 0.0f;
#pragma unroll
    for (int v = 0; v < 8; ++v) {
        int row = tm * 16 + mb + v;
        if (row < M) {
            float* crow = C + outOff + (size_t)row * outStride;
            if (n0 < N) crow[n0] = acc0[v] + bv0;
            if (n1 < N) crow[n1] = acc1[v] + bv1;
        }
    }
}

// ---------------------------------------------------------------------------
// Pack f32 activations -> f16, row stride + column offset, zero-pad K..Kp.
// ---------------------------------------------------------------------------
__global__ void k_pack_a(const float* __restrict__ src, _Float16* __restrict__ dst,
                         int M, int K, int Kp, int stride, int colOff)
{
    size_t i = (size_t)blockIdx.x * blockDim.x + threadIdx.x;
    size_t total = (size_t)M * Kp;
    if (i >= total) return;
    int m = (int)(i / Kp), k = (int)(i % Kp);
    dst[i] = (k < K) ? (_Float16)src[(size_t)m * stride + colOff + k] : (_Float16)0.0f;
}

// Strided weight pack: dst[n*Kp+k] = src[off + n*sN + k*sK]
__global__ void k_pack_w(const float* __restrict__ src, _Float16* __restrict__ dst,
                         int N, int K, int Kp, long sK, long sN, long off)
{
    size_t i = (size_t)blockIdx.x * blockDim.x + threadIdx.x;
    size_t total = (size_t)N * Kp;
    if (i >= total) return;
    int n = (int)(i / Kp), k = (int)(i % Kp);
    dst[i] = (k < K) ? (_Float16)src[off + (long)n * sN + (long)k * sK] : (_Float16)0.0f;
}

// ---------------------------------------------------------------------------
// im2col: x(B,3,img,img) -> rows=(b,hy,wx), cols=(c,p,q), f16 (cols mult of 32)
// ---------------------------------------------------------------------------
__global__ void k_im2col(const float* __restrict__ x, _Float16* __restrict__ out,
                         int Bn, int ps, int hh, int img)
{
    int cols = 3 * ps * ps;
    size_t total = (size_t)Bn * hh * hh * cols;
    size_t i = (size_t)blockIdx.x * blockDim.x + threadIdx.x;
    if (i >= total) return;
    int c = (int)(i % cols);
    int m = (int)(i / cols);
    int b  = m / (hh * hh);
    int t  = m % (hh * hh);
    int hy = t / hh, wx = t % hh;
    int ch = c / (ps * ps);
    int rr = (c % (ps * ps)) / ps;
    int qq = c % ps;
    out[i] = (_Float16)x[(((size_t)b * 3 + ch) * img + hy * ps + rr) * img + wx * ps + qq];
}

// ---------------------------------------------------------------------------
// In-place LayerNorm + SiLU per row (block per row, LDS reduction).
// ---------------------------------------------------------------------------
__global__ __launch_bounds__(256)
void k_ln_silu(float* __restrict__ x, const float* __restrict__ g,
               const float* __restrict__ b, int D)
{
    __shared__ float red[256];
    const int r = blockIdx.x, t = threadIdx.x;
    float* row = x + (size_t)r * D;

    float s = 0.0f;
    for (int c = t; c < D; c += 256) s += row[c];
    red[t] = s; __syncthreads();
    for (int o = 128; o > 0; o >>= 1) { if (t < o) red[t] += red[t + o]; __syncthreads(); }
    const float mean = red[0] / D; __syncthreads();

    float s2 = 0.0f;
    for (int c = t; c < D; c += 256) { float d = row[c] - mean; s2 += d * d; }
    red[t] = s2; __syncthreads();
    for (int o = 128; o > 0; o >>= 1) { if (t < o) red[t] += red[t + o]; __syncthreads(); }
    const float inv = rsqrtf(red[0] / D + 1e-5f);

    for (int c = t; c < D; c += 256) {
        float v = (row[c] - mean) * inv * g[c] + b[c];
        row[c] = siluf(v);
    }
}

// ---------------------------------------------------------------------------
// Rotary concat: out(B,L,D+4) = [in(+add) | sin(l*s), cos(l*s), sin(r*s), cos(r*s)]
// ---------------------------------------------------------------------------
__global__ void k_rotary(const float* __restrict__ in, const float* __restrict__ add,
                         float* __restrict__ out, int Bn, int L, int D, int maxf)
{
    const int Do = D + 4;
    size_t total = (size_t)Bn * L * Do;
    size_t i = (size_t)blockIdx.x * blockDim.x + threadIdx.x;
    if (i >= total) return;
    int c  = (int)(i % Do);
    int bl = (int)(i / Do);
    int l  = bl % L;
    if (c < D) {
        float v = in[(size_t)bl * D + c];
        if (add) v += add[(size_t)bl * D + c];
        out[i] = v;
    } else {
        const float s = 6.2831853071795864769f / (float)maxf;
        float pos = (float)l * s;
        float rw  = (float)(l / maxf) * s;
        float v;
        switch (c - D) {
            case 0:  v = __sinf(pos); break;
            case 1:  v = __cosf(pos); break;
            case 2:  v = __sinf(rw);  break;
            default: v = __cosf(rw);  break;
        }
        out[i] = v;
    }
}

// ---------------------------------------------------------------------------
// Causal depthwise conv1d (K=4) + SiLU. x has row stride xStride (xz buffer).
// ---------------------------------------------------------------------------
__global__ void k_conv_silu(const float* __restrict__ x, int xStride,
                            const float* __restrict__ w, const float* __restrict__ bias,
                            float* __restrict__ out, int Bn, int L, int di)
{
    size_t total = (size_t)Bn * L * di;
    size_t i = (size_t)blockIdx.x * blockDim.x + threadIdx.x;
    if (i >= total) return;
    int d = (int)(i % di);
    int l = (int)((i / di) % L);
    int b = (int)(i / ((size_t)di * L));
    const float* base = x + (size_t)b * L * xStride + d;
    float acc = bias[d];
#pragma unroll
    for (int k = 0; k < 4; ++k) {
        int ls = l + k - 3;
        if (ls >= 0) acc += base[(size_t)ls * xStride] * w[d * 4 + k];
    }
    out[i] = siluf(acc);
}

// softplus in place
__global__ void k_softplus(float* __restrict__ x, size_t n)
{
    size_t i = (size_t)blockIdx.x * blockDim.x + threadIdx.x;
    if (i >= n) return;
    float v = x[i];
    x[i] = (v > 30.0f) ? v : log1pf(__expf(v));
}

// ---------------------------------------------------------------------------
// Selective scan: one thread per (b, d), 16-state recurrence in registers.
// Bm/Cm live in xdbl columns [dtr, dtr+16) and [dtr+16, dtr+32).
// ---------------------------------------------------------------------------
__global__ void k_scan(const float* __restrict__ u, const float* __restrict__ dt,
                       const float* __restrict__ xdbl, int xst, int dtr,
                       const float* __restrict__ A_log, const float* __restrict__ Dp,
                       float* __restrict__ y, int Bn, int L, int di)
{
    int i = blockIdx.x * blockDim.x + threadIdx.x;
    if (i >= Bn * di) return;
    int d = i % di, b = i / di;
    float A[16], h[16];
#pragma unroll
    for (int n = 0; n < 16; ++n) { A[n] = -__expf(A_log[d * 16 + n]); h[n] = 0.0f; }
    const float Dd = Dp[d];
    for (int l = 0; l < L; ++l) {
        size_t r = (size_t)b * L + l;
        float dtv = dt[r * di + d];
        float uv  = u[r * di + d];
        const float* Bm = xdbl + r * xst + dtr;
        const float* Cm = Bm + 16;
        float du = dtv * uv, acc = 0.0f;
#pragma unroll
        for (int n = 0; n < 16; ++n) {
            h[n] = __expf(dtv * A[n]) * h[n] + du * Bm[n];
            acc += h[n] * Cm[n];
        }
        y[r * di + d] = acc + uv * Dd;
    }
}

// out[m*C+c] = a[m*aSt+c] * silu(z[m*zSt+c])
__global__ void k_mul_silu(const float* __restrict__ a, int aSt,
                           const float* __restrict__ z, int zSt,
                           float* __restrict__ out, int M, int C)
{
    size_t total = (size_t)M * C;
    size_t i = (size_t)blockIdx.x * blockDim.x + threadIdx.x;
    if (i >= total) return;
    int m = (int)(i / C), c = (int)(i % C);
    out[i] = a[(size_t)m * aSt + c] * siluf(z[(size_t)m * zSt + c]);
}

__global__ void k_reverse(const float* __restrict__ src, float* __restrict__ dst,
                          int Bn, int L, int D)
{
    size_t total = (size_t)Bn * L * D;
    size_t i = (size_t)blockIdx.x * blockDim.x + threadIdx.x;
    if (i >= total) return;
    int c = (int)(i % D);
    int l = (int)((i / D) % L);
    int b = (int)(i / ((size_t)D * L));
    dst[((size_t)b * L + (L - 1 - l)) * D + c] = src[i];
}

__global__ void k_add_scale(const float* __restrict__ a, const float* __restrict__ b,
                            float* __restrict__ out, size_t n, float s)
{
    size_t i = (size_t)blockIdx.x * blockDim.x + threadIdx.x;
    if (i >= n) return;
    out[i] = (a[i] + b[i]) * s;
}

// bwd_transform input pack: out[(b*196+l)*512 + k*128 + i] = inter1[b, l*4+k, i]
__global__ void k_bst_pack(const float* __restrict__ inter1, _Float16* __restrict__ out)
{
    size_t total = (size_t)8 * 196 * 512;
    size_t i = (size_t)blockIdx.x * blockDim.x + threadIdx.x;
    if (i >= total) return;
    int c = (int)(i % 512);
    int m = (int)(i / 512);
    int b = m / 196, l = m % 196;
    int k = c / 128, ii = c % 128;
    out[i] = (_Float16)inter1[((size_t)b * 784 + (l * 4 + k)) * 132 + ii];
}

// bwd_transform weight pack: W[o][k*128+i] = w[o, i, k]  (w: 256 x 128 x 4)
__global__ void k_bst_wpack(const float* __restrict__ w, _Float16* __restrict__ out)
{
    size_t total = (size_t)256 * 512;
    size_t i = (size_t)blockIdx.x * blockDim.x + threadIdx.x;
    if (i >= total) return;
    int c = (int)(i % 512);
    int n = (int)(i / 512);
    int k = c / 128, ii = c % 128;
    out[i] = (_Float16)w[(size_t)n * 512 + ii * 4 + k];
}

// mean over L then dot with cls_w, per batch
__global__ __launch_bounds__(256)
void k_mean_cls(const float* __restrict__ x, const float* __restrict__ w,
                const float* __restrict__ bb, float* __restrict__ out, int L, int D)
{
    __shared__ float red[256];
    const int b = blockIdx.x, t = threadIdx.x;
    float dot = 0.0f;
    for (int f = t; f < D; f += 256) {
        float s = 0.0f;
        for (int l = 0; l < L; ++l) s += x[((size_t)b * L + l) * D + f];
        dot += (s / (float)L) * w[f];
    }
    red[t] = dot; __syncthreads();
    for (int o = 128; o > 0; o >>= 1) { if (t < o) red[t] += red[t + o]; __syncthreads(); }
    if (t == 0) out[b] = red[0] + bb[0];
}

// ---------------------------------------------------------------------------
// Host-side orchestration
// ---------------------------------------------------------------------------
static inline int ceilDiv(int a, int b) { return (a + b - 1) / b; }
static inline int pad32(int k) { return (k + 31) & ~31; }
static inline int blk1d(size_t n) { return (int)((n + 255) / 256); }

struct MambaW {
    const float *in_proj, *conv_w, *conv_b, *x_proj, *dt_proj,
                *dt_bias, *A_log, *D, *out_proj;
};
struct Scratch {
    _Float16 *xpack, *xcpack, *dtpack, *ypack, *wpack;
    float    *xz, *xcact, *xdbl, *dt, *yscan, *ymul;
};

static void gemm(hipStream_t st, const _Float16* A, const _Float16* W,
                 const float* bias, float* C, int M, int N, int Kp,
                 long outStride, long outOff)
{
    dim3 g(ceilDiv(N, 32), ceilDiv(M, 64));
    k_gemm_wmma<<<g, 128, 0, st>>>(A, W, bias, C, M, N, Kp, outStride, outOff);
}
static void packA(hipStream_t st, const float* s, _Float16* d, int M, int K, int Kp,
                  int stride, int off)
{
    k_pack_a<<<blk1d((size_t)M * Kp), 256, 0, st>>>(s, d, M, K, Kp, stride, off);
}
static void packW(hipStream_t st, const float* s, _Float16* d, int N, int K, int Kp,
                  long sK, long sN, long off)
{
    k_pack_w<<<blk1d((size_t)N * Kp), 256, 0, st>>>(s, d, N, K, Kp, sK, sN, off);
}

static void run_mamba(hipStream_t st, const float* x, float* out,
                      int Bn, int L, int dm, int di, int dtr,
                      const MambaW& w, const Scratch& s)
{
    const int M = Bn * L;
    const int dmp = pad32(dm), dip = pad32(di), dtp = pad32(dtr);
    const int nx = dtr + 32;

    // xz = x @ in_proj^T
    packA(st, x, s.xpack, M, dm, dmp, dm, 0);
    packW(st, w.in_proj, s.wpack, 2 * di, dm, dmp, 1, dm, 0);
    gemm(st, s.xpack, s.wpack, nullptr, s.xz, M, 2 * di, dmp, 2 * di, 0);

    // xc = silu(causal_conv1d(xz[:, :di]))
    k_conv_silu<<<blk1d((size_t)M * di), 256, 0, st>>>(s.xz, 2 * di, w.conv_w, w.conv_b,
                                                      s.xcact, Bn, L, di);
    // xdbl = xc @ x_proj^T
    packA(st, s.xcact, s.xcpack, M, di, dip, di, 0);
    packW(st, w.x_proj, s.wpack, nx, di, dip, 1, di, 0);
    gemm(st, s.xcpack, s.wpack, nullptr, s.xdbl, M, nx, dip, nx, 0);

    // dt = softplus(xdbl[:, :dtr] @ dt_proj^T + dt_bias)
    packA(st, s.xdbl, s.dtpack, M, dtr, dtp, nx, 0);
    packW(st, w.dt_proj, s.wpack, di, dtr, dtp, 1, dtr, 0);
    gemm(st, s.dtpack, s.wpack, w.dt_bias, s.dt, M, di, dtp, di, 0);
    k_softplus<<<blk1d((size_t)M * di), 256, 0, st>>>(s.dt, (size_t)M * di);

    // selective scan
    k_scan<<<blk1d((size_t)Bn * di), 256, 0, st>>>(s.xcact, s.dt, s.xdbl, nx, dtr,
                                                   w.A_log, w.D, s.yscan, Bn, L, di);
    // (y * silu(z)) @ out_proj^T
    k_mul_silu<<<blk1d((size_t)M * di), 256, 0, st>>>(s.yscan, di, s.xz + di, 2 * di,
                                                      s.ymul, M, di);
    packA(st, s.ymul, s.ypack, M, di, dip, di, 0);
    packW(st, w.out_proj, s.wpack, dm, di, dip, 1, di, 0);
    gemm(st, s.ypack, s.wpack, nullptr, out, M, dm, dip, dm, 0);
}

static void run_bidir(hipStream_t st, const float* x, float* out,
                      int Bn, int L, int dm, int di, int dtr,
                      const MambaW& fw, const MambaW& bw,
                      const float* g, const float* beta,
                      const Scratch& s, float* dirA, float* dirB, float* dirC)
{
    const size_t n = (size_t)Bn * L * dm;
    run_mamba(st, x, dirA, Bn, L, dm, di, dtr, fw, s);
    k_reverse<<<blk1d(n), 256, 0, st>>>(x, dirC, Bn, L, dm);
    run_mamba(st, dirC, dirB, Bn, L, dm, di, dtr, bw, s);
    k_reverse<<<blk1d(n), 256, 0, st>>>(dirB, dirC, Bn, L, dm);
    k_add_scale<<<blk1d(n), 256, 0, st>>>(dirA, dirC, out, n, 0.5f);
    k_ln_silu<<<Bn * L, 256, 0, st>>>(out, g, beta, dm);
}

static MambaW readMamba(void* const* d_in, int& p)
{
    MambaW w;
    w.in_proj  = (const float*)d_in[p++];
    w.conv_w   = (const float*)d_in[p++];
    w.conv_b   = (const float*)d_in[p++];
    w.x_proj   = (const float*)d_in[p++];
    w.dt_proj  = (const float*)d_in[p++];
    w.dt_bias  = (const float*)d_in[p++];
    w.A_log    = (const float*)d_in[p++];
    w.D        = (const float*)d_in[p++];
    w.out_proj = (const float*)d_in[p++];
    return w;
}

extern "C" void kernel_launch(void* const* d_in, const int* in_sizes, int n_in,
                              void* d_out, int out_size, void* d_ws, size_t ws_size,
                              hipStream_t stream)
{
    (void)in_sizes; (void)n_in; (void)out_size; (void)ws_size;
    const int Bn = 8, IMG = 224;
    const int L0 = 196, d0 = 256, dm0 = 260, di0 = 520, dtr0 = 17; // patch 16
    const int L1 = 784, d1 = 128, dm1 = 132, di1 = 264, dtr1 = 9;  // patch 8
    const int M0 = Bn * L0, M1 = Bn * L1;

    int p = 0;
    const float* X = (const float*)d_in[p++];
    const float *pe0w = (const float*)d_in[p++], *pe0b = (const float*)d_in[p++],
                *pe0g = (const float*)d_in[p++], *pe0bb = (const float*)d_in[p++];
    const float *pe1w = (const float*)d_in[p++], *pe1b = (const float*)d_in[p++],
                *pe1g = (const float*)d_in[p++], *pe1bb = (const float*)d_in[p++];
    const float *fstw = (const float*)d_in[p++], *fstb = (const float*)d_in[p++],
                *fstg = (const float*)d_in[p++], *fstbb = (const float*)d_in[p++];
    const float *bstw = (const float*)d_in[p++], *bstb = (const float*)d_in[p++],
                *bstg = (const float*)d_in[p++], *bstbb = (const float*)d_in[p++];
    MambaW fm0f = readMamba(d_in, p), fm0b = readMamba(d_in, p);
    const float *fm0g = (const float*)d_in[p++], *fm0beta = (const float*)d_in[p++];
    MambaW fm1f = readMamba(d_in, p), fm1b = readMamba(d_in, p);
    const float *fm1g = (const float*)d_in[p++], *fm1beta = (const float*)d_in[p++];
    MambaW bm0f = readMamba(d_in, p), bm0b = readMamba(d_in, p);
    const float *bm0g = (const float*)d_in[p++], *bm0beta = (const float*)d_in[p++];
    const float *clsw = (const float*)d_in[p++], *clsb = (const float*)d_in[p++];
    float* out = (float*)d_out;

    // ---- workspace bump allocator --------------------------------------
    char* base = (char*)d_ws;
    size_t off = 0;
    auto allocB = [&](size_t bytes) -> void* {
        void* r = base + off;
        off += (bytes + 255) & ~(size_t)255;
        return r;
    };
    auto aF = [&](size_t n) { return (float*)allocB(n * 4); };
    auto aH = [&](size_t n) { return (_Float16*)allocB(n * 2); };

    _Float16* acol0   = aH((size_t)M0 * 768);
    _Float16* acol1   = aH((size_t)M1 * 192);
    float*    pe0y    = aF((size_t)M0 * d0);
    float*    e0      = aF((size_t)M0 * dm0);
    float*    inter0  = aF((size_t)M0 * dm0);
    _Float16* fstpack = aH((size_t)M0 * 256);
    float*    newseq  = aF((size_t)M1 * d1);
    float*    pe1y    = aF((size_t)M1 * d1);
    float*    e1      = aF((size_t)M1 * dm1);
    float*    inter1  = aF((size_t)M1 * dm1);
    _Float16* bstpack = aH((size_t)M0 * 512);
    float*    curpre  = aF((size_t)M0 * d0);
    float*    cur     = aF((size_t)M0 * dm0);
    float*    cur2    = aF((size_t)M0 * dm0);
    float*    finalb  = aF((size_t)M0 * dm0);

    // shared mamba scratch (sized for max over fm0/fm1/bm0)
    const size_t mMdm  = (size_t)M1 * dm1 > (size_t)M0 * dm0 ? (size_t)M1 * dm1 : (size_t)M0 * dm0;
    const size_t mMdmp = (size_t)M1 * pad32(dm1) > (size_t)M0 * pad32(dm0)
                         ? (size_t)M1 * pad32(dm1) : (size_t)M0 * pad32(dm0);
    const size_t mMdi  = (size_t)M1 * di1 > (size_t)M0 * di0 ? (size_t)M1 * di1 : (size_t)M0 * di0;
    const size_t mMdip = (size_t)M1 * pad32(di1) > (size_t)M0 * pad32(di0)
                         ? (size_t)M1 * pad32(di1) : (size_t)M0 * pad32(di0);
    const size_t mMnx  = (size_t)M1 * (dtr1 + 32) > (size_t)M0 * (dtr0 + 32)
                         ? (size_t)M1 * (dtr1 + 32) : (size_t)M0 * (dtr0 + 32);

    Scratch s;
    s.xpack  = aH(mMdmp);
    s.xcpack = aH(mMdip);
    s.dtpack = aH((size_t)M1 * 32 > (size_t)M0 * 32 ? (size_t)M1 * 32 : (size_t)M0 * 32);
    s.wpack  = aH((size_t)(2 * di0) * pad32(dm0));   // largest weight: in_proj fm0
    s.xz     = aF(2 * mMdi);
    s.xcact  = aF(mMdi);
    s.xdbl   = aF(mMnx);
    s.dt     = aF(mMdi);
    s.yscan  = aF(mMdi);
    s.ymul   = s.dt;          // dt consumed by scan before ymul is produced
    s.ypack  = s.xcpack;      // xcpack consumed by x_proj gemm before ypack

    float* dirA = aF(mMdm);
    float* dirB = aF(mMdm);
    float* dirC = aF(mMdm);

    // ---- scale 0: patch embed (16) -> rotary -> bidir fm0 ---------------
    k_im2col<<<blk1d((size_t)M0 * 768), 256, 0, stream>>>(X, acol0, Bn, 16, 14, IMG);
    packW(stream, pe0w, s.wpack, d0, 768, 768, 1, 768, 0);
    gemm(stream, acol0, s.wpack, pe0b, pe0y, M0, d0, 768, d0, 0);
    k_ln_silu<<<M0, 256, 0, stream>>>(pe0y, pe0g, pe0bb, d0);
    k_rotary<<<blk1d((size_t)M0 * dm0), 256, 0, stream>>>(pe0y, nullptr, e0, Bn, L0, d0, 14);
    run_bidir(stream, e0, inter0, Bn, L0, dm0, di0, dtr0, fm0f, fm0b,
              fm0g, fm0beta, s, dirA, dirB, dirC);

    // ---- fwd_transform: inter0[:,:, :256] -> (B, L0*4, 128) -------------
    packA(stream, inter0, fstpack, M0, 256, 256, dm0, 0);
    for (int kk = 0; kk < 4; ++kk) {
        // fst.w (i=256, o=128, k=4): W[o][i] = w[i*512 + o*4 + kk]
        packW(stream, fstw, s.wpack, d1, 256, 256, /*sK=*/512, /*sN=*/4, /*off=*/kk);
        gemm(stream, fstpack, s.wpack, fstb, newseq, M0, d1, 256,
             /*outStride=*/512, /*outOff=*/(long)kk * 128);
    }
    k_ln_silu<<<M1, 256, 0, stream>>>(newseq, fstg, fstbb, d1);

    // ---- scale 1: patch embed (8) + newseq -> rotary -> bidir fm1 -------
    k_im2col<<<blk1d((size_t)M1 * 192), 256, 0, stream>>>(X, acol1, Bn, 8, 28, IMG);
    packW(stream, pe1w, s.wpack, d1, 192, 192, 1, 192, 0);
    gemm(stream, acol1, s.wpack, pe1b, pe1y, M1, d1, 192, d1, 0);
    k_ln_silu<<<M1, 256, 0, stream>>>(pe1y, pe1g, pe1bb, d1);
    k_rotary<<<blk1d((size_t)M1 * dm1), 256, 0, stream>>>(pe1y, newseq, e1, Bn, L1, d1, 28);
    run_bidir(stream, e1, inter1, Bn, L1, dm1, di1, dtr1, fm1f, fm1b,
              fm1g, fm1beta, s, dirA, dirB, dirC);

    // ---- bwd_transform: inter1[:,:, :128] -> (B, L0, 256) ---------------
    k_bst_pack<<<blk1d((size_t)M0 * 512), 256, 0, stream>>>(inter1, bstpack);
    k_bst_wpack<<<blk1d((size_t)256 * 512), 256, 0, stream>>>(bstw, s.wpack);
    gemm(stream, bstpack, s.wpack, bstb, curpre, M0, d0, 512, d0, 0);
    k_ln_silu<<<M0, 256, 0, stream>>>(curpre, bstg, bstbb, d0);
    k_rotary<<<blk1d((size_t)M0 * dm0), 256, 0, stream>>>(curpre, nullptr, cur, Bn, L0, d0, 14);

    // ---- residual + bidir bm0 -> mean + classifier ----------------------
    k_add_scale<<<blk1d((size_t)M0 * dm0), 256, 0, stream>>>(cur, inter0, cur2,
                                                             (size_t)M0 * dm0, 1.0f);
    run_bidir(stream, cur2, finalb, Bn, L0, dm0, di0, dtr0, bm0f, bm0b,
              bm0g, bm0beta, s, dirA, dirB, dirC);
    k_mean_cls<<<Bn, 256, 0, stream>>>(finalb, clsw, clsb, out, L0, dm0);
}
